// TransformerBlockQuantum_65481071400827
// MI455X (gfx1250) — compile-verified, matrix operand order checked by reading, not run on hardware
//
#include <hip/hip_runtime.h>
#include <hip/hip_bf16.h>
#include <math.h>

typedef __attribute__((ext_vector_type(16))) _Float16 v16h;
typedef __attribute__((ext_vector_type(8)))  _Float16 v8h;
typedef __attribute__((ext_vector_type(8)))  float    v8f;

#define S_LEN 2048
#define EMB   512
#define NH    8
#define DK    64
#define FFN   2048
#define NTOK  4096   // B*S = 2*2048
#define LN_EPS 1e-5f

// ---------------------------------------------------------------------------
// Fragment load: A or B fragment of v_wmma_f32_16x16x32_f16 from a row-major
// matrix where frag "rows" (M for A, N for B) are matrix rows and K is
// contiguous. Per ISA 7.12.2: lane<16 -> K {0..7,16..23}, lane>=16 -> K {8..15,24..31}.
// ---------------------------------------------------------------------------
__device__ __forceinline__ v16h frag_ld(const _Float16* tile, int ld) {
  int lane = threadIdx.x & 31;
  int r  = lane & 15;
  int hi = lane >> 4;
  const _Float16* p = tile + (size_t)r * ld + hi * 8;
  v8h lo = *(const v8h*)(p);
  v8h up = *(const v8h*)(p + 16);
  union { v16h v; v8h h[2]; } u;
  u.h[0] = lo; u.h[1] = up;
  return u.v;
}

__device__ __forceinline__ v8f wmma16(v16h a, v16h b, v8f c) {
  return __builtin_amdgcn_wmma_f32_16x16x32_f16(false, a, false, b,
                                                (short)0, c, false, false);
}

// ---------------------------------------------------------------------------
// Kernel 0: f32 -> f16 weight conversion
// ---------------------------------------------------------------------------
__global__ __launch_bounds__(256) void cvt_f16_kernel(const float* s, _Float16* d, int n) {
  int i = blockIdx.x * 256 + threadIdx.x;
  if (i < n) d[i] = (_Float16)s[i];
}

// ---------------------------------------------------------------------------
// Kernel 1: quantum projection. One thread per (b,s,h): cos(x+theta) cumprod,
// element 0 replaced by prod(c[1:]). Writes proj [BH,S,64] and projT [BH,64,S].
// ---------------------------------------------------------------------------
__global__ __launch_bounds__(256) void proj_kernel(const float* __restrict__ x,
                                                   const float* __restrict__ theta,
                                                   _Float16* __restrict__ proj,
                                                   _Float16* __restrict__ projT) {
  int t = blockIdx.x * 256 + threadIdx.x;      // 32768 threads
  int h = t & 7;
  int s = (t >> 3) & (S_LEN - 1);
  int b = t >> 14;
  const float* xp = x + ((size_t)(b * S_LEN + s)) * EMB + h * DK;
  size_t bh = (size_t)b * NH + h;
  _Float16* pr = proj  + (bh * S_LEN + s) * DK;
  _Float16* pT = projT + bh * DK * S_LEN + s;
  float run = 1.f, tail = 1.f;
  #pragma unroll 8
  for (int d = 0; d < DK; ++d) {
    float c = __cosf(xp[d] + theta[d]);
    run *= c;
    if (d >= 1) {
      tail *= c;
      _Float16 v = (_Float16)run;
      pr[d] = v;
      pT[(size_t)d * S_LEN] = v;
    }
  }
  _Float16 z0 = (_Float16)tail;
  pr[0] = z0;
  pT[0] = z0;
}

// ---------------------------------------------------------------------------
// Kernel 2: flash attention per (b,h), TRANSPOSED score tiles.
// Block = 4 waves; each wave owns a 16-query tile.
//   S^T(32 keys x 16 q) = K(32x64) @ Q^T          (A = K rows, B = Q rows)
//   ctx^T(64 d x 16 q) += V^T(64xS) @ P^T         (A = projT rows, B = probs)
// The C-layout of S^T is lane-for-lane identical to the B-fragment layout
// needed for P^T, so probs convert f32->f16 fully in-register.
// Softmax stats are one scalar per lane (its query column); cross-lane
// reduction is a single xor-16 shuffle.
// ---------------------------------------------------------------------------
__global__ __launch_bounds__(128) void attn_kernel(const _Float16* __restrict__ proj,
                                                   const _Float16* __restrict__ projT,
                                                   _Float16* __restrict__ ctx_out) {
  const int lane = threadIdx.x & 31;
  const int wave = threadIdx.x >> 5;           // 0..3
  const int bh   = blockIdx.x >> 5;            // 32 blocks per (b,h)
  const int qb   = (blockIdx.x & 31) * 64 + wave * 16;
  const int b    = bh >> 3;
  const int h    = bh & 7;

  const _Float16* P  = proj  + (size_t)bh * S_LEN * DK;
  const _Float16* PT = projT + (size_t)bh * DK * S_LEN;

  // Q fragments as B operand (queries = N columns), pre-scaled by 1/sqrt(64)
  union { v16h v; _Float16 e[16]; } bq0, bq1;
  bq0.v = frag_ld(P + (size_t)qb * DK + 0,  DK);
  bq1.v = frag_ld(P + (size_t)qb * DK + 32, DK);
  #pragma unroll
  for (int i = 0; i < 16; ++i) {
    bq0.e[i] = (_Float16)((float)bq0.e[i] * 0.125f);
    bq1.e[i] = (_Float16)((float)bq1.e[i] * 0.125f);
  }

  const v8f vzero = {0.f,0.f,0.f,0.f,0.f,0.f,0.f,0.f};
  v8f accT[4]; accT[0]=vzero; accT[1]=vzero; accT[2]=vzero; accT[3]=vzero;
  float m = -INFINITY, l = 0.f;

  for (int kb = 0; kb < S_LEN / 32; ++kb) {
    const int t0 = kb * 32;

    // ---- S^T tiles: s0 = keys t0..t0+15, s1 = keys t0+16..t0+31 ----
    v8f s0 = vzero, s1 = vzero;
    v16h ak;
    ak = frag_ld(P + (size_t)(t0     ) * DK + 0,  DK); s0 = wmma16(ak, bq0.v, s0);
    ak = frag_ld(P + (size_t)(t0     ) * DK + 32, DK); s0 = wmma16(ak, bq1.v, s0);
    ak = frag_ld(P + (size_t)(t0 + 16) * DK + 0,  DK); s1 = wmma16(ak, bq0.v, s1);
    ak = frag_ld(P + (size_t)(t0 + 16) * DK + 32, DK); s1 = wmma16(ak, bq1.v, s1);

    // ---- online softmax: one scalar per lane (its query column) ----
    float vmax = s0[0];
    #pragma unroll
    for (int r = 1; r < 8; ++r) vmax = fmaxf(vmax, s0[r]);
    #pragma unroll
    for (int r = 0; r < 8; ++r) vmax = fmaxf(vmax, s1[r]);
    vmax = fmaxf(vmax, __shfl_xor(vmax, 16, 32));   // combine key halves
    float mn   = fmaxf(m, vmax);
    float corr = __expf(m - mn);
    m = mn;

    // exp + in-register conversion to B fragment of P^T
    union { v16h v; _Float16 e[16]; } bp;
    float rs = 0.f;
    #pragma unroll
    for (int r = 0; r < 8; ++r) {
      float p = __expf(s0[r] - mn);
      bp.e[r] = (_Float16)p;
      rs += p;
    }
    #pragma unroll
    for (int r = 0; r < 8; ++r) {
      float p = __expf(s1[r] - mn);
      bp.e[8 + r] = (_Float16)p;
      rs += p;
    }
    rs += __shfl_xor(rs, 16, 32);
    l = l * corr + rs;

    // rescale ctx^T accumulators (corr is column-uniform -> plain scalar mul)
    #pragma unroll
    for (int n = 0; n < 4; ++n)
      #pragma unroll
      for (int r = 0; r < 8; ++r)
        accT[n][r] *= corr;

    // ---- ctx^T += V^T(d x 32keys) @ P^T(32keys x 16q) ----
    #pragma unroll
    for (int n = 0; n < 4; ++n) {
      v16h av = frag_ld(PT + (size_t)(n * 16) * S_LEN + t0, S_LEN);
      accT[n] = wmma16(av, bp.v, accT[n]);
    }
  }

  // ---- normalize; per lane the 8 rows are contiguous d-values of one token ----
  const float linv = 1.f / l;
  const int q  = lane & 15;
  const int hi = lane >> 4;
  const size_t tok = (size_t)b * S_LEN + qb + q;
  #pragma unroll
  for (int n = 0; n < 4; ++n) {
    v8h o;
    #pragma unroll
    for (int r = 0; r < 8; ++r) o[r] = (_Float16)(accT[n][r] * linv);
    *(v8h*)(ctx_out + tok * EMB + h * DK + n * 16 + 8 * hi) = o;
  }
}

// ---------------------------------------------------------------------------
// Kernel 3/6: out[M=4096, N=512] = A(f16)[M,K] @ W(f16)[N,K]^T + bias + resid.
// Block = 8 waves = 64x128 tile; wave = 16x64; K-loop in steps of 32.
// ---------------------------------------------------------------------------
__global__ __launch_bounds__(256) void gemm_res_kernel(const _Float16* __restrict__ A,
                                                       const _Float16* __restrict__ W,
                                                       const float* __restrict__ bias,
                                                       const float* __restrict__ resid,
                                                       float* __restrict__ out,
                                                       int K, int ldA, int ldB) {
  const int lane = threadIdx.x & 31;
  const int wave = threadIdx.x >> 5;           // 0..7
  const int bx = blockIdx.x & 3;               // 4 col-blocks of 128
  const int by = blockIdx.x >> 2;              // 64 row-blocks of 64
  const int row0 = by * 64 + (wave & 3) * 16;
  const int col0 = bx * 128 + (wave >> 2) * 64;

  const v8f vzero = {0.f,0.f,0.f,0.f,0.f,0.f,0.f,0.f};
  v8f acc[4]; acc[0]=vzero; acc[1]=vzero; acc[2]=vzero; acc[3]=vzero;

  for (int k = 0; k < K; k += 32) {
    v16h a = frag_ld(A + (size_t)row0 * ldA + k, ldA);
    #pragma unroll
    for (int n = 0; n < 4; ++n) {
      v16h b = frag_ld(W + (size_t)(col0 + n * 16) * ldB + k, ldB);
      acc[n] = wmma16(a, b, acc[n]);
    }
  }

  const int col = lane & 15;
  const int hi  = lane >> 4;
  #pragma unroll
  for (int n = 0; n < 4; ++n) {
    int e = col0 + n * 16 + col;
    float bv = bias[e];
    #pragma unroll
    for (int r = 0; r < 8; ++r) {
      int row = row0 + r + 8 * hi;
      size_t idx = (size_t)row * EMB + e;
      out[idx] = acc[n][r] + bv + resid[idx];
    }
  }
}

// ---------------------------------------------------------------------------
// LayerNorm over rows of 512 (one block per row, 256 threads x 2 elems)
// ---------------------------------------------------------------------------
__global__ __launch_bounds__(256) void ln_kernel(const float* __restrict__ in,
                                                 const float* __restrict__ g,
                                                 const float* __restrict__ b,
                                                 float* __restrict__ out) {
  __shared__ float s_sum[256], s_sq[256];
  int row = blockIdx.x;
  const float* p = in + (size_t)row * EMB;
  float v0 = p[threadIdx.x];
  float v1 = p[threadIdx.x + 256];
  s_sum[threadIdx.x] = v0 + v1;
  s_sq[threadIdx.x]  = v0 * v0 + v1 * v1;
  __syncthreads();
  for (int st = 128; st > 0; st >>= 1) {
    if (threadIdx.x < st) {
      s_sum[threadIdx.x] += s_sum[threadIdx.x + st];
      s_sq[threadIdx.x]  += s_sq[threadIdx.x + st];
    }
    __syncthreads();
  }
  float mu  = s_sum[0] * (1.f / EMB);
  float var = s_sq[0] * (1.f / EMB) - mu * mu;
  float rs  = rsqrtf(var + LN_EPS);
  float* o = out + (size_t)row * EMB;
  o[threadIdx.x]       = (v0 - mu) * rs * g[threadIdx.x]       + b[threadIdx.x];
  o[threadIdx.x + 256] = (v1 - mu) * rs * g[threadIdx.x + 256] + b[threadIdx.x + 256];
}

// ---------------------------------------------------------------------------
// Kernel 5: quantum FFN hidden: hid[tok,j] = relu(b1[j] + sum_k q_k W1[j,k]),
// q_k = cos(x1[tok,k]) * cos(phi[k]), k<8. Each thread: 8 j's -> one b128 store.
// ---------------------------------------------------------------------------
__global__ __launch_bounds__(256) void hid_kernel(const float* __restrict__ x1,
                                                  const float* __restrict__ W1,
                                                  const float* __restrict__ b1,
                                                  const float* __restrict__ phi,
                                                  _Float16* __restrict__ hid) {
  int tok = blockIdx.x;
  float q[8];
  #pragma unroll
  for (int k = 0; k < 8; ++k)
    q[k] = __cosf(x1[(size_t)tok * EMB + k]) * __cosf(phi[k]);
  int j0 = threadIdx.x * 8;
  v8h o;
  #pragma unroll
  for (int jj = 0; jj < 8; ++jj) {
    int j = j0 + jj;
    float s = b1[j];
    #pragma unroll
    for (int k = 0; k < 8; ++k) s += q[k] * W1[j * 8 + k];
    o[jj] = (_Float16)fmaxf(s, 0.f);
  }
  *(v8h*)(hid + (size_t)tok * FFN + j0) = o;
}

// ---------------------------------------------------------------------------
// Host launcher
// ---------------------------------------------------------------------------
extern "C" void kernel_launch(void* const* d_in, const int* in_sizes, int n_in,
                              void* d_out, int out_size, void* d_ws, size_t ws_size,
                              hipStream_t stream) {
  (void)in_sizes; (void)n_in; (void)out_size; (void)ws_size;
  const float* x        = (const float*)d_in[0];
  const float* theta    = (const float*)d_in[1];
  const float* phi      = (const float*)d_in[2];
  const float* W_comb   = (const float*)d_in[3];
  const float* b_comb   = (const float*)d_in[4];
  const float* W1       = (const float*)d_in[5];
  const float* b1       = (const float*)d_in[6];
  const float* W2       = (const float*)d_in[7];
  const float* b2       = (const float*)d_in[8];
  const float* ln1_g    = (const float*)d_in[9];
  const float* ln1_b    = (const float*)d_in[10];
  const float* ln2_g    = (const float*)d_in[11];
  const float* ln2_b    = (const float*)d_in[12];
  float* outp = (float*)d_out;

  // workspace carve-up
  char* w = (char*)d_ws;
  _Float16* proj   = (_Float16*)w;                       w += (size_t)16 * S_LEN * DK * 2;   //  4 MB
  _Float16* projT  = (_Float16*)w;                       w += (size_t)16 * DK * S_LEN * 2;   //  4 MB
  _Float16* ctx16  = (_Float16*)w;                       w += (size_t)NTOK * EMB * 2;        //  4 MB
  _Float16* Wc16   = (_Float16*)w;                       w += (size_t)EMB * EMB * 2;         // 0.5 MB
  _Float16* W216   = (_Float16*)w;                       w += (size_t)EMB * FFN * 2;         //  2 MB
  _Float16* hid16  = (_Float16*)w;                       w += (size_t)NTOK * FFN * 2;        // 16 MB
  float*    y1     = (float*)w;                          w += (size_t)NTOK * EMB * 4;        //  8 MB
  float*    x1     = (float*)w;                          w += (size_t)NTOK * EMB * 4;        //  8 MB
  float*    y2     = (float*)w;                          w += (size_t)NTOK * EMB * 4;        //  8 MB

  // 0) weight conversion
  cvt_f16_kernel<<<(EMB * EMB + 255) / 256, 256, 0, stream>>>(W_comb, Wc16, EMB * EMB);
  cvt_f16_kernel<<<(EMB * FFN + 255) / 256, 256, 0, stream>>>(W2, W216, EMB * FFN);

  // 1) quantum projection
  proj_kernel<<<128, 256, 0, stream>>>(x, theta, proj, projT);

  // 2) flash attention (16 bh-pairs x 32 query-blocks)
  attn_kernel<<<512, 128, 0, stream>>>(proj, projT, ctx16);

  // 3) output projection + residual, then LN1
  gemm_res_kernel<<<256, 256, 0, stream>>>(ctx16, Wc16, b_comb, x, y1,
                                           /*K=*/EMB, /*ldA=*/EMB, /*ldB=*/EMB);
  ln_kernel<<<NTOK, 256, 0, stream>>>(y1, ln1_g, ln1_b, x1);

  // 4) quantum FFN hidden
  hid_kernel<<<NTOK, 256, 0, stream>>>(x1, W1, b1, phi, hid16);

  // 5) FFN down-projection + residual, then LN2 -> output
  gemm_res_kernel<<<256, 256, 0, stream>>>(hid16, W216, b2, x1, y2,
                                           /*K=*/FFN, /*ldA=*/FFN, /*ldB=*/FFN);
  ln_kernel<<<NTOK, 256, 0, stream>>>(y2, ln2_g, ln2_b, outp);
}
